// GAT_71846212928267
// MI455X (gfx1250) — matrix operand order repaired and checked by reading, not compiled, not run on hardware
//
#include <hip/hip_runtime.h>

// GAT HeteroConv (6 relations, shared params) for MI455X / gfx1250.
// GEMMs: v_wmma_f32_16x16x32_bf16, 16x16 tile per wave32, K-step 32.
// Edge phase: wave-per-edge gather/scatter with global_atomic_add_f32;
// segment-max via int-max/uint-min float-atomic trick.

typedef __attribute__((ext_vector_type(16))) __bf16 bf16x16;
typedef __attribute__((ext_vector_type(8)))  __bf16 bf16x8;
typedef __attribute__((ext_vector_type(8)))  float  f32x8;

#define F_IN 512      // input features (= K of GEMM1 and GEMM2)
#define HC   512      // H*C concat dim
#define NOUT 128      // final output dim

__device__ __forceinline__ void atomic_max_f32(float* addr, float val) {
  // works for mixed-sign values when initialized to -inf
  if (val >= 0.0f) atomicMax((int*)addr, __float_as_int(val));
  else             atomicMin((unsigned int*)addr, __float_as_uint(val));
}

__device__ __forceinline__ float leaky(float a) { return a > 0.0f ? a : 0.2f * a; }

// ---------------- conversion / init kernels ----------------
__global__ void k_cvt_x(const float* __restrict__ x, __bf16* __restrict__ xb, int total) {
  int t = blockIdx.x * 256 + threadIdx.x;
  if (t < total) xb[t] = (__bf16)x[t];
}

// WbT[n*512+k] = concat(W_src,W_dst)(k, n)  -- transposed so B-fragments are contiguous
__global__ void k_cvt_w(const float* __restrict__ Wsrc, const float* __restrict__ Wdst,
                        __bf16* __restrict__ WbT) {
  int t = blockIdx.x * 256 + threadIdx.x;        // over 1024*512
  if (t >= 1024 * F_IN) return;
  int n = t >> 9, k = t & (F_IN - 1);
  float v = (n < HC) ? Wsrc[k * HC + n] : Wdst[k * HC + (n - HC)];
  WbT[t] = (__bf16)v;
}

__global__ void k_cvt_wlin(const float* __restrict__ Wlin, __bf16* __restrict__ WlbT) {
  int t = blockIdx.x * 256 + threadIdx.x;        // over 128*512
  if (t >= NOUT * HC) return;
  int n = t >> 9, k = t & (HC - 1);
  WlbT[t] = (__bf16)Wlin[k * NOUT + n];
}

__global__ void k_zero(float* __restrict__ p, int total) {
  int t = blockIdx.x * 256 + threadIdx.x;
  if (t < total) p[t] = 0.0f;
}

__global__ void k_init_rel(float* __restrict__ amax, float* __restrict__ den, int total) {
  int t = blockIdx.x * 256 + threadIdx.x;
  if (t < total) { amax[t] = -__builtin_huge_valf(); den[t] = 0.0f; }
}

// ---------------- GEMM1: P = x @ [W_src | W_dst]  (N x 1024) ----------------
// Columns <512 stored as xs (f32, needed for messages); columns >=512 are the
// xd half and are immediately reduced against att_dst into a_d (never stored).
__global__ void k_gemm_proj(const __bf16* __restrict__ xb, const __bf16* __restrict__ WbT,
                            const float* __restrict__ att_dst,
                            float* __restrict__ xs, float* __restrict__ a_d,
                            int tiles_m) {
  const int lane = threadIdx.x & 31;
  const int tile = blockIdx.x * 8 + (threadIdx.x >> 5);
  const int tm = tile >> 6;                       // tiles_n = 64
  const int tn = tile & 63;
  if (tm >= tiles_m) return;
  const int rowA = tm * 16 + (lane & 15);
  const int colB = tn * 16 + (lane & 15);
  const int kbA = (lane < 16) ? 0 : 16;           // A K-base: contiguous k = 16h + s, not AMD table order
  const int kbB = (lane < 16) ? 0 : 16;           // B K-base: same contiguous k = 16h + s as A
  const __bf16* ap = xb  + (size_t)rowA * F_IN;
  const __bf16* bp = WbT + (size_t)colB * F_IN;
  f32x8 c = {};
  for (int k0 = 0; k0 < F_IN; k0 += 32) {
    bf16x8  alo = *(const bf16x8*)(ap + k0 + kbA);
    bf16x8  ahi = *(const bf16x8*)(ap + k0 + kbA + 8);
    bf16x16 b   = *(const bf16x16*)(bp + k0 + kbB);
    bf16x16 a;
#pragma unroll
    for (int i = 0; i < 8; ++i) { a[i] = alo[i]; a[i + 8] = ahi[i]; }
    c = __builtin_amdgcn_wmma_f32_16x16x32_bf16(false, a, false, b, (short)0, c,
                                                false, false);
  }
  const int mb = tm * 16 + ((lane < 16) ? 0 : 8); // C/D layout: rows mb..mb+7
  if (colB < HC) {
#pragma unroll
    for (int v = 0; v < 8; ++v) xs[(size_t)(mb + v) * HC + colB] = c[v];
  } else {
    const int j = colB - HC;
    const int h = j >> 8, cc = j & 255;
    const float wgt = att_dst[h * 256 + cc];
#pragma unroll
    for (int v = 0; v < 8; ++v) atomicAdd(&a_d[(mb + v) * 2 + h], c[v] * wgt);
  }
}

// a_s[n,h] = sum_c xs[n,h,c] * att_src[h,c]
__global__ void k_a_src(const float* __restrict__ xs, const float* __restrict__ att_src,
                        float* __restrict__ a_s, int n) {
  int t = blockIdx.x * 256 + threadIdx.x;         // over N*2
  if (t >= n * 2) return;
  int node = t >> 1, h = t & 1;
  const float* xp = xs + (size_t)node * HC + h * 256;
  const float* apt = att_src + h * 256;
  float s = 0.0f;
  for (int c = 0; c < 256; c += 4) {
    float4 xv = *(const float4*)(xp + c);
    float4 av = *(const float4*)(apt + c);
    s += xv.x * av.x + xv.y * av.y + xv.z * av.z + xv.w * av.w;
  }
  a_s[t] = s;
}

// ---------------- per-relation edge kernels ----------------
// edges e in [0,E): (src,dst) from ei; e in [E,E+N): self loop (e-E, e-E)
__global__ void k_alpha(const long long* __restrict__ ei, int E, int n,
                        const float* __restrict__ a_s, const float* __restrict__ a_d,
                        float* __restrict__ alpha, float* __restrict__ amax) {
  int e = blockIdx.x * 256 + threadIdx.x;
  if (e >= E + n) return;
  int src, dst;
  if (e < E) { src = (int)ei[e]; dst = (int)ei[E + e]; }
  else       { src = dst = e - E; }
#pragma unroll
  for (int h = 0; h < 2; ++h) {
    float a = leaky(a_s[src * 2 + h] + a_d[dst * 2 + h]);
    alpha[e * 2 + h] = a;
    atomic_max_f32(&amax[dst * 2 + h], a);
  }
}

__global__ void k_exden(const long long* __restrict__ ei, int E, int n,
                        const float* __restrict__ amax,
                        float* __restrict__ alpha /* in: alpha, out: exp */,
                        float* __restrict__ den) {
  int e = blockIdx.x * 256 + threadIdx.x;
  if (e >= E + n) return;
  int dst = (e < E) ? (int)ei[E + e] : (e - E);
#pragma unroll
  for (int h = 0; h < 2; ++h) {
    float ex = __expf(alpha[e * 2 + h] - amax[dst * 2 + h]);
    alpha[e * 2 + h] = ex;
    atomicAdd(&den[dst * 2 + h], ex);
  }
}

// wave32 per edge: lane covers 8 contiguous channels per head; scatter-add f32
__global__ void k_scatter(const long long* __restrict__ ei, int E, int n,
                          const float* __restrict__ ex, const float* __restrict__ den,
                          const float* __restrict__ xs, float* __restrict__ acc) {
  const int lane = threadIdx.x & 31;
  const int e = blockIdx.x * 8 + (threadIdx.x >> 5);
  if (e >= E + n) return;
  int src, dst;
  if (e < E) { src = (int)ei[e]; dst = (int)ei[E + e]; }
  else       { src = dst = e - E; }
#pragma unroll
  for (int h = 0; h < 2; ++h) {
    const float w = ex[e * 2 + h] / (den[dst * 2 + h] + 1e-16f);
    const int base = h * 256 + lane * 8;
    const float* sp = xs + (size_t)src * HC + base;
    float* apd = acc + (size_t)dst * HC + base;
    float4 v0 = *(const float4*)(sp);
    float4 v1 = *(const float4*)(sp + 4);
    atomicAdd(apd + 0, v0.x * w); atomicAdd(apd + 1, v0.y * w);
    atomicAdd(apd + 2, v0.z * w); atomicAdd(apd + 3, v0.w * w);
    atomicAdd(apd + 4, v1.x * w); atomicAdd(apd + 5, v1.y * w);
    atomicAdd(apd + 6, v1.z * w); atomicAdd(apd + 7, v1.w * w);
  }
}

// accb = bf16(acc + 6*bias_conv)  (bias added once per relation in reference)
__global__ void k_acc2bf(const float* __restrict__ acc, const float* __restrict__ bias,
                         __bf16* __restrict__ accb, int total) {
  int t = blockIdx.x * 256 + threadIdx.x;
  if (t >= total) return;
  int col = t & (HC - 1);
  accb[t] = (__bf16)(acc[t] + 6.0f * bias[col]);
}

// ---------------- GEMM2: out = accb @ W_lin + b_lin  (N x 128) ----------------
__global__ void k_gemm_lin(const __bf16* __restrict__ accb, const __bf16* __restrict__ WlbT,
                           const float* __restrict__ b_lin, float* __restrict__ out,
                           int tiles_m) {
  const int lane = threadIdx.x & 31;
  const int tile = blockIdx.x * 8 + (threadIdx.x >> 5);
  const int tm = tile >> 3;                       // tiles_n = 8
  const int tn = tile & 7;
  if (tm >= tiles_m) return;
  const int rowA = tm * 16 + (lane & 15);
  const int colB = tn * 16 + (lane & 15);
  const int kbA = (lane < 16) ? 0 : 16;
  const int kbB = (lane < 16) ? 0 : 16;
  const __bf16* ap = accb + (size_t)rowA * HC;
  const __bf16* bp = WlbT + (size_t)colB * HC;
  f32x8 c = {};
  for (int k0 = 0; k0 < HC; k0 += 32) {
    bf16x8  alo = *(const bf16x8*)(ap + k0 + kbA);
    bf16x8  ahi = *(const bf16x8*)(ap + k0 + kbA + 8);
    bf16x16 b   = *(const bf16x16*)(bp + k0 + kbB);
    bf16x16 a;
#pragma unroll
    for (int i = 0; i < 8; ++i) { a[i] = alo[i]; a[i + 8] = ahi[i]; }
    c = __builtin_amdgcn_wmma_f32_16x16x32_bf16(false, a, false, b, (short)0, c,
                                                false, false);
  }
  const int mb = tm * 16 + ((lane < 16) ? 0 : 8);
  const float bl = b_lin[colB];
#pragma unroll
  for (int v = 0; v < 8; ++v) out[(size_t)(mb + v) * NOUT + colB] = c[v] + bl;
}

// ---------------- host launcher ----------------
extern "C" void kernel_launch(void* const* d_in, const int* in_sizes, int n_in,
                              void* d_out, int out_size, void* d_ws, size_t ws_size,
                              hipStream_t stream) {
  (void)n_in; (void)out_size; (void)ws_size;
  const float* x       = (const float*)d_in[0];
  const long long* ei[6];
  for (int r = 0; r < 6; ++r) ei[r] = (const long long*)d_in[1 + r];   // int64 per reference
  const float* W_src   = (const float*)d_in[7];
  const float* W_dst   = (const float*)d_in[8];
  const float* att_src = (const float*)d_in[9];
  const float* att_dst = (const float*)d_in[10];
  const float* bias_c  = (const float*)d_in[11];
  const float* W_lin   = (const float*)d_in[12];
  const float* b_lin   = (const float*)d_in[13];
  float* out = (float*)d_out;

  const int N = in_sizes[0] / F_IN;               // 20000
  const int tiles_m = N / 16;                     // N is a multiple of 16

  // workspace carve-up (256B aligned)
  char* base = (char*)d_ws;
  size_t off = 0;
  auto carve = [&](size_t bytes) -> char* {
    char* p = base + off;
    off = (off + bytes + 255) & ~(size_t)255;
    return p;
  };
  __bf16* xb   = (__bf16*)carve((size_t)N * F_IN * 2);   // also reused as accb later
  __bf16* WbT  = (__bf16*)carve((size_t)1024 * F_IN * 2);
  __bf16* WlbT = (__bf16*)carve((size_t)NOUT * HC * 2);
  float*  xs   = (float*)carve((size_t)N * HC * 4);
  float*  a_s  = (float*)carve((size_t)N * 2 * 4);
  float*  a_d  = (float*)carve((size_t)N * 2 * 4);
  float*  acc  = (float*)carve((size_t)N * HC * 4);
  float*  amax = (float*)carve((size_t)N * 2 * 4);
  float*  den  = (float*)carve((size_t)N * 2 * 4);
  // per-relation edge buffer sized for the largest relation
  int Emax = 0;
  for (int r = 0; r < 6; ++r) { int Er = in_sizes[1 + r] / 2; if (Er > Emax) Emax = Er; }
  float*  exb  = (float*)carve((size_t)(Emax + N) * 2 * 4);
  __bf16* accb = xb;                              // alias: xb dead after GEMM1

  const int T = 256;
  auto blocks = [](long long n, int t) { return (unsigned)((n + t - 1) / t); };

  // conversions + zero init
  k_cvt_x   <<<blocks((long long)N * F_IN, T), T, 0, stream>>>(x, xb, N * F_IN);
  k_cvt_w   <<<blocks(1024 * F_IN, T),          T, 0, stream>>>(W_src, W_dst, WbT);
  k_cvt_wlin<<<blocks(NOUT * HC, T),            T, 0, stream>>>(W_lin, WlbT);
  k_zero    <<<blocks((long long)N * 2, T),     T, 0, stream>>>(a_d, N * 2);
  k_zero    <<<blocks((long long)N * HC, T),    T, 0, stream>>>(acc, N * HC);

  // fused projection GEMM (xs stored, a_d reduced in epilogue)
  k_gemm_proj<<<blocks((long long)tiles_m * 64, 8), T, 0, stream>>>(
      xb, WbT, att_dst, xs, a_d, tiles_m);

  // a_s from xs
  k_a_src<<<blocks((long long)N * 2, T), T, 0, stream>>>(xs, att_src, a_s, N);

  // 6 relations: segment softmax + message scatter into acc
  for (int r = 0; r < 6; ++r) {
    const int E = in_sizes[1 + r] / 2;
    const int EA = E + N;
    k_init_rel<<<blocks((long long)N * 2, T), T, 0, stream>>>(amax, den, N * 2);
    k_alpha   <<<blocks(EA, T), T, 0, stream>>>(ei[r], E, N, a_s, a_d, exb, amax);
    k_exden   <<<blocks(EA, T), T, 0, stream>>>(ei[r], E, N, amax, exb, den);
    k_scatter <<<blocks(EA, 8), T, 0, stream>>>(ei[r], E, N, exb, den, xs, acc);
  }

  // fold 6*bias, convert to bf16, final WMMA GEMM with b_lin epilogue
  k_acc2bf  <<<blocks((long long)N * HC, T), T, 0, stream>>>(acc, bias_c, accb, N * HC);
  k_gemm_lin<<<blocks((long long)tiles_m * 8, 8), T, 0, stream>>>(
      accb, WlbT, b_lin, out, tiles_m);
}
